// NetworkPairsTopologyModel_6528350290171
// MI455X (gfx1250) — matrix operand order repaired and checked by reading, not compiled, not run on hardware
//
#include <hip/hip_runtime.h>
#include <hip/hip_bf16.h>
#include <stdint.h>

#define HDIM    128
#define NNODES  10000
#define EOLD    320000
#define ENEW    320000
#define KDEC    288           // 264 padded to multiple of 32
#define HASH_BITS 20
#define HASH_CAP  (1u << HASH_BITS)
#define HASH_MASK (HASH_CAP - 1u)
#define EMPTY_KEY 0xFFFFFFFFu
#define LN_EPS  1e-5f
#define AGG_EPS 1e-6f
#define MTILES_PER_WAVE 8

typedef __attribute__((ext_vector_type(16))) __bf16 v16bf;
typedef __attribute__((ext_vector_type(8)))  float  v8f;
typedef __attribute__((ext_vector_type(4)))  unsigned int u32x4;

union Frag {
  v16bf v;
  u32x4 q[2];
  unsigned int u[8];
  unsigned short h[16];
};

__device__ __forceinline__ unsigned short f2bf(float f) {
  unsigned int u = __float_as_uint(f);
  unsigned int r = u + 0x7FFFu + ((u >> 16) & 1u);
  return (unsigned short)(r >> 16);
}

__device__ __forceinline__ float wave_sum(float v) {
#pragma unroll
  for (int off = 16; off > 0; off >>= 1) v += __shfl_xor(v, off, 32);
  return v;
}

// ---------------------------------------------------------------- fills
__global__ void fill_f32_kernel(float* p, float v, int n) {
  int i = blockIdx.x * blockDim.x + threadIdx.x;
  if (i < n) p[i] = v;
}
__global__ void fill_u16_kernel(unsigned short* p, unsigned short v, int n) {
  int i = blockIdx.x * blockDim.x + threadIdx.x;
  if (i < n) p[i] = v;
}
__global__ void fill_u32_kernel(unsigned int* p, unsigned int v, int n) {
  int i = blockIdx.x * blockDim.x + threadIdx.x;
  if (i < n) p[i] = v;
}
__global__ void copy_u16_kernel(const unsigned short* __restrict__ s,
                                unsigned short* __restrict__ d, int n) {
  int i = blockIdx.x * blockDim.x + threadIdx.x;
  if (i < n) d[i] = s[i];
}

// ---------------- weight fp32 -> bf16, pre-swizzled into WMMA B-fragment order
// dst[((kt*tilesN + nt)*32 + lane)*16 + s] = B[kt*32 + 16*(lane>>4) + s][nt*16 + (lane&15)]
__global__ void cvt_w_swz_kernel(const float* __restrict__ src, unsigned short* __restrict__ dst,
                                 int K, int N, int Kpad) {
  int idx = blockIdx.x * blockDim.x + threadIdx.x;
  if (idx >= Kpad * N) return;
  int s    = idx & 15;
  int lane = (idx >> 4) & 31;
  int t    = idx >> 9;
  int tilesN = N >> 4;
  int nt = t % tilesN;
  int kt = t / tilesN;
  int k   = kt * 32 + ((lane >> 4) << 4) + s;
  int col = (nt << 4) + (lane & 15);
  dst[idx] = (k < K) ? f2bf(src[(size_t)k * N + col]) : (unsigned short)0;
}

// bias2[j] = fusion.b[j] + sum_i fusion.w_top[i][j]   (x_init == ones)
__global__ void fusion_bias_kernel(const float* __restrict__ fw, const float* __restrict__ fb,
                                   float* __restrict__ out) {
  int t = threadIdx.x;  // 128
  float s = fb[t];
  for (int i = 0; i < HDIM; ++i) s += fw[(size_t)i * HDIM + t];
  out[t] = s;
}

// old_feats = concat(edge_attr_old, flow_old)  (E x 4)
__global__ void concat_old_kernel(const float* __restrict__ attr, const float* __restrict__ flow,
                                  float* __restrict__ outf, int E) {
  int e = blockIdx.x * blockDim.x + threadIdx.x;
  if (e >= E) return;
  outf[(size_t)e * 4 + 0] = attr[(size_t)e * 3 + 0];
  outf[(size_t)e * 4 + 1] = attr[(size_t)e * 3 + 1];
  outf[(size_t)e * 4 + 2] = attr[(size_t)e * 3 + 2];
  outf[(size_t)e * 4 + 3] = flow[e];
}

// ------------------------------------------------ small dense proj (din<=8)
__global__ void small_proj_kernel(const float* __restrict__ in, int din,
                                  const float* __restrict__ W, const float* __restrict__ bias,
                                  float* __restrict__ out, unsigned short* __restrict__ outb,
                                  int E) {
  __shared__ float sW[8 * HDIM];
  __shared__ float sB[HDIM];
  int t = threadIdx.x;  // 128
  for (int i = t; i < din * HDIM; i += HDIM) sW[i] = W[i];
  sB[t] = bias[t];
  __syncthreads();
  int e0 = blockIdx.x * 64;
  for (int r = 0; r < 64; ++r) {
    int e = e0 + r;
    if (e >= E) break;
    float acc = sB[t];
    for (int i = 0; i < din; ++i) acc += in[(size_t)e * din + i] * sW[i * HDIM + t];
    out[(size_t)e * HDIM + t] = acc;
    if (outb) outb[(size_t)e * HDIM + t] = f2bf(acc);
  }
}

// ------------------------------------------------ WMMA GEMM (bf16 in, f32 acc)
// One wave owns one N-tile + a chunk of M tiles; B fragments (pre-swizzled)
// stay in registers across the chunk. All A fragments for a tile are loaded
// first (one load clause), then KSTEPS WMMAs run back-to-back.
template <int KSTEPS, int RELU, int HASB>
__global__ void wmma_gemm_kernel(const unsigned short* __restrict__ A,
                                 const unsigned short* __restrict__ Bsw,
                                 const float* __restrict__ bias,
                                 float* __restrict__ C,
                                 unsigned short* __restrict__ Cb,
                                 int M, int N) {
  const int K = KSTEPS * 32;
  int wave = (blockIdx.x * blockDim.x + threadIdx.x) >> 5;
  int lane = threadIdx.x & 31;
  int tilesN = N >> 4;
  int tilesM = (M + 15) >> 4;
  int nChunks = (tilesM + MTILES_PER_WAVE - 1) / MTILES_PER_WAVE;
  if (wave >= tilesN * nChunks) return;
  int nt = wave % tilesN;
  int mt0 = (wave / tilesN) * MTILES_PER_WAVE;
  int half = lane >> 4;
  int l15 = lane & 15;
  int col = (nt << 4) + l15;

  Frag bf[KSTEPS];
#pragma unroll
  for (int ks = 0; ks < KSTEPS; ++ks) {
    const u32x4* p = (const u32x4*)(Bsw + ((size_t)(ks * tilesN + nt) * 32 + lane) * 16);
    bf[ks].q[0] = p[0];
    bf[ks].q[1] = p[1];
  }
  float bv = bias[col];

  int mtEnd = mt0 + MTILES_PER_WAVE;
  if (mtEnd > tilesM) mtEnd = tilesM;
  for (int mt = mt0; mt < mtEnd; ++mt) {
    int mBase = mt << 4;
    int mA = mBase + l15;
    if (mA >= M) mA = M - 1;
    const unsigned short* Arow = A + (size_t)mA * K;

    Frag a[KSTEPS];
#pragma unroll
    for (int ks = 0; ks < KSTEPS; ++ks) {
      a[ks].q[0] = *(const u32x4*)(Arow + ks * 32 + half * 16);       // slots 0..7
      a[ks].q[1] = *(const u32x4*)(Arow + ks * 32 + half * 16 + 8);  // slots 8..15
    }
    v8f acc;
#pragma unroll
    for (int r = 0; r < 8; ++r) acc[r] = bv;
#pragma unroll
    for (int ks = 0; ks < KSTEPS; ++ks)
      acc = __builtin_amdgcn_wmma_f32_16x16x32_bf16(false, a[ks].v, false, bf[ks].v,
                                                    (short)0, acc, false, false);

    int mRow = mBase + (half ? 8 : 0);
    float* Crow = C + (size_t)mRow * N + col;
    if (mBase + 16 <= M) {  // fast path: full tile, no predication
#pragma unroll
      for (int r = 0; r < 8; ++r) {
        float vv = acc[r];
        if (RELU) vv = vv > 0.f ? vv : 0.f;
        Crow[(size_t)r * N] = vv;
      }
      if (HASB) {
        unsigned short* Brow = Cb + (size_t)mRow * N + col;
#pragma unroll
        for (int r = 0; r < 8; ++r) {
          float vv = acc[r];
          if (RELU) vv = vv > 0.f ? vv : 0.f;
          Brow[(size_t)r * N] = f2bf(vv);
        }
      }
    } else {
#pragma unroll
      for (int r = 0; r < 8; ++r) {
        int m = mRow + r;
        if (m < M) {
          float vv = acc[r];
          if (RELU) vv = vv > 0.f ? vv : 0.f;
          C[(size_t)m * N + col] = vv;
          if (HASB) Cb[(size_t)m * N + col] = f2bf(vv);
        }
      }
    }
  }
}

// ---------------- e_hat = Ce + Dx[dst] + Ex[src]; sigma; atomic segment sums
__global__ void edge_assemble_kernel(float* __restrict__ ehat,
                                     const float* __restrict__ Dx, const float* __restrict__ Ex,
                                     const float* __restrict__ Bx,
                                     const int* __restrict__ src, const int* __restrict__ dst,
                                     float* __restrict__ num, float* __restrict__ den, int E) {
  int wave = (blockIdx.x * blockDim.x + threadIdx.x) >> 5;
  int lane = threadIdx.x & 31;
  if (wave >= E) return;
  int s = src[wave], d = dst[wave];
  int c = lane * 4;
  size_t eo = (size_t)wave * HDIM + c;
  size_t so = (size_t)s * HDIM + c;
  size_t doff = (size_t)d * HDIM + c;
#pragma unroll
  for (int i = 0; i < 4; ++i) {
    float eh = ehat[eo + i] + Dx[doff + i] + Ex[so + i];
    ehat[eo + i] = eh;
    float sig = 1.f / (1.f + __expf(-eh));
    atomicAdd(&num[doff + i], sig * Bx[so + i]);
    atomicAdd(&den[doff + i], sig);
  }
}

// ---------------- x += relu(LN(Ax + num/(den+eps)));  also emit bf16 copy
__global__ void node_update_kernel(const float* __restrict__ Ax, const float* __restrict__ num,
                                   const float* __restrict__ den,
                                   const float* __restrict__ g, const float* __restrict__ b,
                                   float* __restrict__ x, unsigned short* __restrict__ xb, int n) {
  int wave = (blockIdx.x * blockDim.x + threadIdx.x) >> 5;
  int lane = threadIdx.x & 31;
  if (wave >= n) return;
  int c = lane * 4;
  size_t o = (size_t)wave * HDIM + c;
  float v[4];
  float s = 0.f, sq = 0.f;
#pragma unroll
  for (int i = 0; i < 4; ++i) {
    v[i] = Ax[o + i] + num[o + i] / (den[o + i] + AGG_EPS);
    s += v[i];
    sq += v[i] * v[i];
  }
  s = wave_sum(s);
  sq = wave_sum(sq);
  float mean = s * (1.f / HDIM);
  float var = sq * (1.f / HDIM) - mean * mean;
  float inv = rsqrtf(var + LN_EPS);
#pragma unroll
  for (int i = 0; i < 4; ++i) {
    float y = (v[i] - mean) * inv * g[c + i] + b[c + i];
    y = y > 0.f ? y : 0.f;
    float xn = x[o + i] + y;
    x[o + i] = xn;
    xb[o + i] = f2bf(xn);
  }
}

// ---------------- e += relu(LN(e_hat)); emit bf16 copy
__global__ void edge_ln_kernel(const float* __restrict__ ehat,
                               const float* __restrict__ g, const float* __restrict__ b,
                               float* __restrict__ e, unsigned short* __restrict__ eb, int E) {
  int wave = (blockIdx.x * blockDim.x + threadIdx.x) >> 5;
  int lane = threadIdx.x & 31;
  if (wave >= E) return;
  int c = lane * 4;
  size_t o = (size_t)wave * HDIM + c;
  float v[4];
  float s = 0.f, sq = 0.f;
#pragma unroll
  for (int i = 0; i < 4; ++i) {
    v[i] = ehat[o + i];
    s += v[i];
    sq += v[i] * v[i];
  }
  s = wave_sum(s);
  sq = wave_sum(sq);
  float mean = s * (1.f / HDIM);
  float var = sq * (1.f / HDIM) - mean * mean;
  float inv = rsqrtf(var + LN_EPS);
#pragma unroll
  for (int i = 0; i < 4; ++i) {
    float y = (v[i] - mean) * inv * g[c + i] + b[c + i];
    y = y > 0.f ? y : 0.f;
    float en = e[o + i] + y;
    e[o + i] = en;
    eb[o + i] = f2bf(en);
  }
}

// ---------------- hash table: build on old edges, probe+align for new edges
__global__ void hash_build_kernel(const int* __restrict__ src, const int* __restrict__ dst,
                                  unsigned int* __restrict__ keys, int* __restrict__ vals, int E) {
  int e = blockIdx.x * blockDim.x + threadIdx.x;
  if (e >= E) return;
  unsigned int key = (unsigned int)src[e] * (unsigned int)NNODES + (unsigned int)dst[e];
  unsigned int h = (key * 2654435761u) >> (32 - HASH_BITS);
  for (;;) {
    unsigned int old = atomicCAS(&keys[h & HASH_MASK], EMPTY_KEY, key);
    if (old == EMPTY_KEY || old == key) { vals[h & HASH_MASK] = e; break; }
    ++h;
  }
}

__global__ void probe_align_kernel(const int* __restrict__ src, const int* __restrict__ dst,
                                   const unsigned int* __restrict__ keys,
                                   const int* __restrict__ vals,
                                   const float* __restrict__ oldfeat,
                                   const float* __restrict__ attr_new,
                                   float* __restrict__ aligned, int E) {
  int e = blockIdx.x * blockDim.x + threadIdx.x;
  if (e >= E) return;
  unsigned int key = (unsigned int)src[e] * (unsigned int)NNODES + (unsigned int)dst[e];
  unsigned int h = (key * 2654435761u) >> (32 - HASH_BITS);
  int match = -1;
  for (unsigned int probe = 0; probe < HASH_CAP; ++probe, ++h) {
    unsigned int k = keys[h & HASH_MASK];
    if (k == EMPTY_KEY) break;
    if (k == key) { match = vals[h & HASH_MASK]; break; }
  }
  bool retained = match >= 0;
  size_t ao = (size_t)e * 8;
#pragma unroll
  for (int i = 0; i < 4; ++i)
    aligned[ao + i] = retained ? oldfeat[(size_t)match * 4 + i] : 0.f;
#pragma unroll
  for (int i = 0; i < 3; ++i) aligned[ao + 4 + i] = attr_new[(size_t)e * 3 + i];
  aligned[ao + 7] = retained ? 0.f : 1.f;
}

// ---------------- decoder: gather [x[src], x[dst], aligned, pad] as bf16 rows
__global__ void edge_repr_kernel(const unsigned short* __restrict__ xb,
                                 const float* __restrict__ aligned,
                                 const int* __restrict__ src, const int* __restrict__ dst,
                                 unsigned short* __restrict__ Adec, long long total) {
  long long idx = (long long)blockIdx.x * blockDim.x + threadIdx.x;
  if (idx >= total) return;
  int e = (int)(idx / KDEC);
  int k = (int)(idx % KDEC);
  unsigned short v;
  if (k < 128)       v = xb[(size_t)src[e] * HDIM + k];
  else if (k < 256)  v = xb[(size_t)dst[e] * HDIM + (k - 128)];
  else if (k < 264)  v = f2bf(aligned[(size_t)e * 8 + (k - 256)]);
  else               v = 0;
  Adec[idx] = v;
}

__global__ void dec2_kernel(const float* __restrict__ h, const float* __restrict__ w,
                            const float* __restrict__ b, float* __restrict__ out, int E) {
  int wave = (blockIdx.x * blockDim.x + threadIdx.x) >> 5;
  int lane = threadIdx.x & 31;
  if (wave >= E) return;
  float s = 0.f;
  for (int i = lane; i < HDIM; i += 32) s += h[(size_t)wave * HDIM + i] * w[i];
  s = wave_sum(s);
  if (lane == 0) out[wave] = s + b[0];
}

// ============================================================ host side
static inline int cdiv(long long a, long long b) { return (int)((a + b - 1) / b); }

extern "C" void kernel_launch(void* const* d_in, const int* in_sizes, int n_in,
                              void* d_out, int out_size, void* d_ws, size_t ws_size,
                              hipStream_t stream) {
  (void)in_sizes; (void)n_in; (void)out_size; (void)ws_size;
  const int n = NNODES;

  // ---- inputs (setup_inputs order; params pytree = sorted dict keys)
  const int*   ei_old   = (const int*)d_in[0];
  const float* attr_old = (const float*)d_in[1];
  const float* flow_old = (const float*)d_in[2];
  const int*   ei_new   = (const int*)d_in[3];
  const float* attr_new = (const float*)d_in[4];
  // d_in[5] = total_nodes (scalar, known constant)
  auto F = [&](int i) { return (const float*)d_in[i]; };
  const float* dec1_b = F(6);
  const float* dec1_w = F(7);
  const float* dec2_b = F(8);
  const float* dec2_w = F(9);
  const float* enc_b  = F(10);
  const float* enc_w  = F(11);
  const float* fus_b  = F(54);
  const float* fus_w  = F(55);
  const float* rsn_b  = F(56);
  const float* rsn_w  = F(57);
  // per-layer leaves: A.b,A.w,B.b,B.w,C.b,C.w,D.b,D.w,E.b,E.w,ln_e.b,ln_e.g,ln_x.b,ln_x.g
  auto layer_base = [&](int stack, int l) { return (stack == 0 ? 12 : 58) + 14 * l; };

  const int* src_o = ei_old;           const int* dst_o = ei_old + EOLD;
  const int* src_n = ei_new;           const int* dst_n = ei_new + ENEW;

  // ---- workspace carve
  char* base = (char*)d_ws;
  size_t off = 0;
  auto alloc = [&](size_t bytes) -> void* {
    void* p = base + off;
    off += (bytes + 255) & ~(size_t)255;
    return p;
  };
  float*          x_f   = (float*)alloc((size_t)n * HDIM * 4);
  unsigned short* x_b   = (unsigned short*)alloc((size_t)n * HDIM * 2);
  unsigned short* hold_b= (unsigned short*)alloc((size_t)n * HDIM * 2);
  float*          Ax    = (float*)alloc((size_t)n * HDIM * 4);
  float*          Bx    = (float*)alloc((size_t)n * HDIM * 4);
  float*          Dx    = (float*)alloc((size_t)n * HDIM * 4);
  float*          Exb   = (float*)alloc((size_t)n * HDIM * 4);
  float*          numb  = (float*)alloc((size_t)n * HDIM * 4);
  float*          denb  = (float*)alloc((size_t)n * HDIM * 4);
  float*          bias2 = (float*)alloc(HDIM * 4);
  float*          oldft = (float*)alloc((size_t)EOLD * 4 * 4);
  float*          align8= (float*)alloc((size_t)ENEW * 8 * 4);
  float*          e_f   = (float*)alloc((size_t)EOLD * HDIM * 4);
  unsigned short* e_b   = (unsigned short*)alloc((size_t)EOLD * HDIM * 2);
  float*          ehat  = (float*)alloc((size_t)EOLD * HDIM * 4);
  unsigned int*   hkeys = (unsigned int*)alloc((size_t)HASH_CAP * 4);
  int*            hvals = (int*)alloc((size_t)HASH_CAP * 4);
  unsigned short* Wb[30];  // 6 layers x {A,B,C,D,E}, swizzled bf16
  for (int i = 0; i < 30; ++i) Wb[i] = (unsigned short*)alloc((size_t)HDIM * HDIM * 2);
  unsigned short* Wfus  = (unsigned short*)alloc((size_t)HDIM * HDIM * 2);
  unsigned short* Wdec1 = (unsigned short*)alloc((size_t)KDEC * HDIM * 2);
  // aliases over dead edge scratch (used only after all GatedGCN layers)
  unsigned short* Adec  = (unsigned short*)e_f;   // E*288 bf16 fits in e_f + e_b
  float*          hdec  = ehat;                    // E*128 f32

  const int TB = 256;
  // ---- convert weights to swizzled bf16
  for (int stack = 0; stack < 2; ++stack)
    for (int l = 0; l < 3; ++l) {
      int lb = layer_base(stack, l);
      for (int j = 0; j < 5; ++j) {
        cvt_w_swz_kernel<<<cdiv(HDIM * HDIM, TB), TB, 0, stream>>>(
            F(lb + 2 * j + 1), Wb[(stack * 3 + l) * 5 + j], HDIM, HDIM, HDIM);
      }
    }
  cvt_w_swz_kernel<<<cdiv(HDIM * HDIM, TB), TB, 0, stream>>>(fus_w + HDIM * HDIM, Wfus,
                                                             HDIM, HDIM, HDIM);
  cvt_w_swz_kernel<<<cdiv(KDEC * HDIM, TB), TB, 0, stream>>>(dec1_w, Wdec1, 264, HDIM, KDEC);
  fusion_bias_kernel<<<1, 128, 0, stream>>>(fus_w, fus_b, bias2);

  // ---- GEMM launch helpers (1 wave = 1 N-tile x MTILES_PER_WAVE M-tiles)
  auto gemm_blocks = [&](int M, int N) {
    int tilesN = N / 16;
    int tilesM = (M + 15) / 16;
    return cdiv(tilesN * cdiv(tilesM, MTILES_PER_WAVE), 4);
  };
  // plain: K=128, no relu, no bf16 out
  auto gemm128 = [&](const unsigned short* A, const unsigned short* W, const float* bias,
                     float* C, int M) {
    wmma_gemm_kernel<4, 0, 0><<<gemm_blocks(M, HDIM), 128, 0, stream>>>(
        A, W, bias, C, nullptr, M, HDIM);
  };

  // ---- encoder input: e = enc_proj(concat(attr_old, flow_old)), x = ones
  concat_old_kernel<<<cdiv(EOLD, TB), TB, 0, stream>>>(attr_old, flow_old, oldft, EOLD);
  small_proj_kernel<<<cdiv(EOLD, 64), 128, 0, stream>>>(oldft, 4, enc_w, enc_b, e_f, e_b, EOLD);
  fill_f32_kernel<<<cdiv((long long)n * HDIM, TB), TB, 0, stream>>>(x_f, 1.0f, n * HDIM);
  fill_u16_kernel<<<cdiv((long long)n * HDIM, TB), TB, 0, stream>>>(x_b, (unsigned short)0x3F80,
                                                                    n * HDIM);

  // ---- GatedGCN stacks
  auto run_stack = [&](int stack, const int* src, const int* dst, int E) {
    for (int l = 0; l < 3; ++l) {
      int lb = layer_base(stack, l);
      unsigned short** W = &Wb[(stack * 3 + l) * 5];
      gemm128(x_b, W[0], F(lb + 0), Ax, n);    // A
      gemm128(x_b, W[1], F(lb + 2), Bx, n);    // B
      gemm128(x_b, W[3], F(lb + 6), Dx, n);    // D
      gemm128(x_b, W[4], F(lb + 8), Exb, n);   // E
      gemm128(e_b, W[2], F(lb + 4), ehat, E);  // C(e)
      fill_f32_kernel<<<cdiv((long long)n * HDIM, TB), TB, 0, stream>>>(numb, 0.f, n * HDIM);
      fill_f32_kernel<<<cdiv((long long)n * HDIM, TB), TB, 0, stream>>>(denb, 0.f, n * HDIM);
      edge_assemble_kernel<<<cdiv((long long)E * 32, TB), TB, 0, stream>>>(
          ehat, Dx, Exb, Bx, src, dst, numb, denb, E);
      node_update_kernel<<<cdiv((long long)n * 32, TB), TB, 0, stream>>>(
          Ax, numb, denb, F(lb + 13), F(lb + 12), x_f, x_b, n);
      edge_ln_kernel<<<cdiv((long long)E * 32, TB), TB, 0, stream>>>(
          ehat, F(lb + 11), F(lb + 10), e_f, e_b, E);
    }
  };
  run_stack(0, src_o, dst_o, EOLD);

  // ---- edge matching (hash) + aligned features
  fill_u32_kernel<<<cdiv(HASH_CAP, TB), TB, 0, stream>>>(hkeys, EMPTY_KEY, HASH_CAP);
  hash_build_kernel<<<cdiv(EOLD, TB), TB, 0, stream>>>(src_o, dst_o, hkeys, hvals, EOLD);
  probe_align_kernel<<<cdiv(ENEW, TB), TB, 0, stream>>>(src_n, dst_n, hkeys, hvals,
                                                        oldft, attr_new, align8, ENEW);

  // ---- node fusion: x = relu(h_old @ W_bot + (b + colsum(W_top)))  [relu + bf16 out]
  copy_u16_kernel<<<cdiv((long long)n * HDIM, TB), TB, 0, stream>>>(x_b, hold_b, n * HDIM);
  wmma_gemm_kernel<4, 1, 1><<<gemm_blocks(n, HDIM), 128, 0, stream>>>(
      hold_b, Wfus, bias2, x_f, x_b, n, HDIM);

  // ---- rsn stack
  small_proj_kernel<<<cdiv(ENEW, 64), 128, 0, stream>>>(align8, 8, rsn_w, rsn_b, e_f, e_b, ENEW);
  run_stack(1, src_n, dst_n, ENEW);

  // ---- decoder
  long long total = (long long)ENEW * KDEC;
  edge_repr_kernel<<<cdiv(total, TB), TB, 0, stream>>>(x_b, align8, src_n, dst_n, Adec, total);
  wmma_gemm_kernel<9, 1, 0><<<gemm_blocks(ENEW, HDIM), 128, 0, stream>>>(
      Adec, Wdec1, dec1_b, hdec, nullptr, ENEW, HDIM);
  dec2_kernel<<<cdiv((long long)ENEW * 32, TB), TB, 0, stream>>>(hdec, dec2_w, dec2_b,
                                                                 (float*)d_out, ENEW);
}
